// PointerGeneratorDecoder_67276367725250
// MI455X (gfx1250) — matrix operand order repaired and checked by reading, not compiled, not run on hardware
//
#include <hip/hip_runtime.h>
#include <cstdint>

// ---------------------------------------------------------------------------
// Pointer-generator decoder for MI455X (gfx1250, wave32, WMMA).
//
// Roofline: per step we must stream enc_proj [64,1024,512] and encoder_hy
// [64,1024,512]. In f32 that is 268 MB/step (26.8 GB total, > 192MB L2).
// We keep both as bf16 (67 MB each): streamed bytes halve AND the 134 MB
// working set becomes L2-resident on MI455X's 192 MB L2, so the 100-step
// loop streams from L2 instead of HBM. All GEMMs use
// v_wmma_f32_16x16x32_bf16 with f32 accumulation; the 3.35G tanh
// evaluations in attention scoring use the hardware v_tanh_f32 TRANS op.
// ---------------------------------------------------------------------------

typedef __attribute__((ext_vector_type(16))) __bf16 bf16x16;
typedef __attribute__((ext_vector_type(8)))  float  f32x8;
using u16 = unsigned short;

#define DEV __device__ __forceinline__

constexpr int Bb   = 64;
constexpr int Tt   = 100;
constexpr int Ss   = 1024;
constexpr int INs  = 256;
constexpr int TRG  = 512;
constexpr int ENC  = 512;
constexpr int Gg   = 2048;   // 4*TRG
constexpr int KCAT = 1280;   // IN + TRG + TRG

DEV float bf2f(u16 x) {
    union { unsigned u; float f; } v; v.u = ((unsigned)x) << 16; return v.f;
}
DEV u16 f2bf(float f) {
    union { float f; unsigned u; } v; v.f = f;
    unsigned r = v.u + 0x7FFFu + ((v.u >> 16) & 1u);
    return (u16)(r >> 16);
}
DEV float sigm(float x) { return 1.f / (1.f + __expf(-x)); }

// CDNA5 has a hardware v_tanh_f32 transcendental; use it when the builtin is
// exposed (hot path: 33.5M tanh per decode step in attention scoring).
DEV float fast_tanh(float x) {
#if __has_builtin(__builtin_amdgcn_tanhf)
    return __builtin_amdgcn_tanhf(x);
#else
    return tanhf(x);
#endif
}

union FragB16 {
    bf16x16 v;
    u16     u[16];
    uint4   q[2];
};

// Wave-level 16x16 tile GEMM: C[m0:m0+16, n0:n0+16] += A[M,K] * W[N,K]^T.
// A row-major (lda = K stride), W row-major [N x K] so B[k][n] = W[n][k].
// A-fragment (16x32 bf16): lane = half*16+l16; row = m0+l16;
//   u[0..7]  = K[k0 + half*8     .. +8)
//   u[8..15] = K[k0 + 16 + half*8 .. +8)   (ISA 7.12.2 16-bit A layout)
// B-fragment (32x16 bf16): col n0+l16; same K layout as the A-fragment.
DEV f32x8 wmma_bf16_AWt(const u16* __restrict__ A, int lda,
                        const u16* __restrict__ W, int ldw,
                        int m0, int n0, int K) {
    const int lane = threadIdx.x & 31;
    const int half = lane >> 4;
    const int l16  = lane & 15;
    f32x8 acc = {0.f, 0.f, 0.f, 0.f, 0.f, 0.f, 0.f, 0.f};
    const u16* pa = A + (m0 + l16) * lda + half * 8;
    const u16* pb = W + (n0 + l16) * ldw + half * 8;
    for (int k0 = 0; k0 < K; k0 += 32) {
        FragB16 fa, fb;
        fa.q[0] = *(const uint4*)(pa + k0);
        fa.q[1] = *(const uint4*)(pa + k0 + 16);
        fb.q[0] = *(const uint4*)(pb + k0);
        fb.q[1] = *(const uint4*)(pb + k0 + 16);
        if (k0 + 64 < K) {
            // locality 3 -> WGP-scope global_prefetch_b8 (pull into all cache
            // levels); operands are L2-resident weights/activations.
            __builtin_prefetch(pa + k0 + 64, 0, 3);
            __builtin_prefetch(pb + k0 + 64, 0, 3);
        }
        acc = __builtin_amdgcn_wmma_f32_16x16x32_bf16(
            false, fa.v, false, fb.v, (short)0, acc, false, false);
    }
    return acc;
}

// ---------------------------------------------------------------------------
// One-time kernels
// ---------------------------------------------------------------------------

// State init: state lives in the output slots (hT/cT/haT/paT) so final-step
// values are the outputs for free.
__global__ void k_init(const float* __restrict__ h0, const float* __restrict__ c0,
                       const float* __restrict__ hattn, const float* __restrict__ pattn,
                       const float* __restrict__ pdehy,
                       float* o_h, float* o_c, float* o_ha, float* o_pa,
                       float* o_pd, float* o_loss) {
    int i = blockIdx.x * blockDim.x + threadIdx.x;
    const int N1 = Bb * TRG;        // 32768
    if (i < N1)                 { o_h[i] = h0[i]; return; }
    i -= N1;
    if (i < N1)                 { o_c[i] = c0[i]; return; }
    i -= N1;
    if (i < N1)                 { o_ha[i] = hattn[i]; return; }
    i -= N1;
    if (i < Bb * Ss)            { o_pa[i] = pattn[i]; return; }
    i -= Bb * Ss;
    if (i < N1)                 { o_pd[i] = pdehy[i]; return; }
    i -= N1;
    if (i == 0)                 { o_loss[0] = 0.f; }
}

// f32 -> bf16 conversions (weights concatenated for the fused gate GEMM).
// Non-temporal loads: one-shot pass, keep L2 clean for resident bf16 tensors.
__global__ void k_convert(const float* __restrict__ enc,
                          const float* __restrict__ wih, const float* __restrict__ whh,
                          const float* __restrict__ wen, const float* __restrict__ wde,
                          const float* __restrict__ wao,
                          u16* enc_bf, u16* wcat, u16* wenb, u16* wdeb, u16* waob) {
    int i = blockIdx.x * blockDim.x + threadIdx.x;
    const int NE = Bb * Ss * ENC;                 // 33,554,432
    if (i < NE) { enc_bf[i] = f2bf(__builtin_nontemporal_load(enc + i)); return; }
    i -= NE;
    const int NW = Gg * KCAT;                     // 2,621,440
    if (i < NW) {
        int g = i / KCAT, k = i - g * KCAT;
        float v = (k < INs + TRG) ? __builtin_nontemporal_load(wih + g * (INs + TRG) + k)
                                  : __builtin_nontemporal_load(whh + g * TRG + (k - (INs + TRG)));
        wcat[i] = f2bf(v);
        return;
    }
    i -= NW;
    if (i < TRG * ENC) { wenb[i] = f2bf(__builtin_nontemporal_load(wen + i)); return; }
    i -= TRG * ENC;
    if (i < TRG * TRG) { wdeb[i] = f2bf(__builtin_nontemporal_load(wde + i)); return; }
    i -= TRG * TRG;
    if (i < TRG * (ENC + TRG)) { waob[i] = f2bf(__builtin_nontemporal_load(wao + i)); }
}

// enc_proj[b,s,:] = encoder_hy[b,s,:] @ w_en.T + b_en, stored bf16.
// M = 65536 (4096 tiles), N = 512 (32 tiles), K = 512.
__global__ void k_encproj(const u16* __restrict__ encb, const u16* __restrict__ wenb,
                          const float* __restrict__ ben, u16* __restrict__ ep) {
    int wid = (blockIdx.x * blockDim.x + threadIdx.x) >> 5;
    int mt = wid >> 5, nt = wid & 31;
    int m0 = mt * 16, n0 = nt * 16;
    f32x8 acc = wmma_bf16_AWt(encb, ENC, wenb, ENC, m0, n0, ENC);
    const int lane = threadIdx.x & 31, half = lane >> 4, l16 = lane & 15;
    float bias = ben[n0 + l16];
#pragma unroll
    for (int r = 0; r < 8; ++r)
        ep[(m0 + half * 8 + r) * TRG + n0 + l16] = f2bf(acc[r] + bias);
}

// ---------------------------------------------------------------------------
// Per-step kernels
// ---------------------------------------------------------------------------

// xcat[b] = [x_t(256) | ha(512) | h(512)] as bf16 (A operand of gate GEMM).
__global__ void k_stagex(const float* __restrict__ input_, const float* __restrict__ ha,
                         const float* __restrict__ h, u16* __restrict__ xcat, int t) {
    int i = blockIdx.x * blockDim.x + threadIdx.x;
    if (i >= Bb * KCAT) return;
    int b = i / KCAT, k = i - b * KCAT;
    float v = (k < INs)        ? input_[(b * Tt + t) * INs + k]
            : (k < INs + TRG)  ? ha[b * TRG + (k - INs)]
                               : h[b * TRG + (k - INs - TRG)];
    xcat[i] = f2bf(v);
}

// gates[64,2048] = xcat @ [w_ih|w_hh]^T + b_ih + b_hh.  512 tiles, K=1280.
__global__ void k_gates(const u16* __restrict__ xcat, const u16* __restrict__ wcat,
                        const float* __restrict__ bih, const float* __restrict__ bhh,
                        float* __restrict__ gates) {
    int wid = (blockIdx.x * blockDim.x + threadIdx.x) >> 5;
    int mt = wid >> 7, nt = wid & 127;
    int m0 = mt * 16, n0 = nt * 16;
    f32x8 acc = wmma_bf16_AWt(xcat, KCAT, wcat, KCAT, m0, n0, KCAT);
    const int lane = threadIdx.x & 31, half = lane >> 4, l16 = lane & 15;
    float bias = bih[n0 + l16] + bhh[n0 + l16];
#pragma unroll
    for (int r = 0; r < 8; ++r)
        gates[(m0 + half * 8 + r) * Gg + n0 + l16] = acc[r] + bias;
}

// LSTM cell elementwise; updates h/c state in the output slots; stages bf16 h.
__global__ void k_cell(const float* __restrict__ gates, float* __restrict__ h,
                       float* __restrict__ c, u16* __restrict__ hbf,
                       u16* __restrict__ cat2) {
    int i = blockIdx.x * blockDim.x + threadIdx.x;
    if (i >= Bb * TRG) return;
    int b = i / TRG, j = i - b * TRG;
    const float* g = gates + b * Gg;
    float ig = sigm(g[j]);
    float fg = sigm(g[TRG + j]);
    float gg = fast_tanh(g[2 * TRG + j]);
    float og = sigm(g[3 * TRG + j]);
    float cn = fg * c[i] + ig * gg;
    float hn = og * fast_tanh(cn);
    c[i] = cn;
    h[i] = hn;
    u16 hb = f2bf(hn);
    hbf[i] = hb;
    cat2[b * (ENC + TRG) + TRG + j] = hb;   // [c_enc | h_new] concat, h half
}

// dh[64,512] = h_new @ w_de^T.  128 tiles, K=512.
__global__ void k_dh(const u16* __restrict__ hbf, const u16* __restrict__ wdeb,
                     float* __restrict__ dh) {
    int wid = (blockIdx.x * blockDim.x + threadIdx.x) >> 5;
    int mt = wid >> 5, nt = wid & 31;
    int m0 = mt * 16, n0 = nt * 16;
    f32x8 acc = wmma_bf16_AWt(hbf, TRG, wdeb, TRG, m0, n0, TRG);
    const int lane = threadIdx.x & 31, half = lane >> 4, l16 = lane & 15;
#pragma unroll
    for (int r = 0; r < 8; ++r)
        dh[(m0 + half * 8 + r) * TRG + n0 + l16] = acc[r];
}

// ee[b,s] = sum_h tanh(enc_proj[b,s,h] + dh[b,h] + pa[b,s]*cv[h]) * warp[h].
// Wave per (b,s); enc_proj streamed bf16 (L2-resident), dh/cv/warp in LDS.
__global__ void k_attnee(const u16* __restrict__ ep, const float* __restrict__ dh,
                         const float* __restrict__ wcv, const float* __restrict__ wwarp,
                         const float* __restrict__ pa, float* __restrict__ ee) {
    const int b = blockIdx.x >> 7;          // 128 chunks of 8 s per b
    const int chunk = blockIdx.x & 127;
    const int tid = threadIdx.x;
    __shared__ float s_dh[TRG], s_cv[TRG], s_wp[TRG];
    for (int j = tid; j < TRG; j += 256) {
        s_dh[j] = dh[b * TRG + j];
        s_cv[j] = wcv[j];
        s_wp[j] = wwarp[j];
    }
    __syncthreads();
    const int wave = tid >> 5, lane = tid & 31;
    const int s = chunk * 8 + wave;
    const u16* row = ep + ((size_t)b * Ss + s) * TRG + lane * 16;
    float pav = pa[b * Ss + s];
    FragB16 f;
    f.q[0] = *(const uint4*)(row);
    f.q[1] = *(const uint4*)(row + 8);
    float acc = 0.f;
#pragma unroll
    for (int i = 0; i < 16; ++i) {
        int hh = lane * 16 + i;
        float x = bf2f(f.u[i]) + s_dh[hh] + pav * s_cv[hh];
        acc += fast_tanh(x) * s_wp[hh];
    }
#pragma unroll
    for (int m = 16; m > 0; m >>= 1) acc += __shfl_xor(acc, m);
    if (lane == 0) ee[b * Ss + s] = acc;
}

// Softmax over S per batch row; writes attn into the attns[t] output slab and
// accumulates pa += attn.
__global__ void k_softmax(const float* __restrict__ ee, float* __restrict__ arow_base,
                          float* __restrict__ pa) {
    const int b = blockIdx.x, tid = threadIdx.x;
    const float* e = ee + b * Ss;
    float* arow = arow_base + b * Ss;
    __shared__ float red[256];
    float m = -1e30f;
    for (int s = tid; s < Ss; s += 256) m = fmaxf(m, e[s]);
    red[tid] = m; __syncthreads();
    for (int off = 128; off > 0; off >>= 1) {
        if (tid < off) red[tid] = fmaxf(red[tid], red[tid + off]);
        __syncthreads();
    }
    float M = red[0]; __syncthreads();
    float sum = 0.f;
    for (int s = tid; s < Ss; s += 256) {
        float v = __expf(e[s] - M);
        arow[s] = v;
        sum += v;
    }
    red[tid] = sum; __syncthreads();
    for (int off = 128; off > 0; off >>= 1) {
        if (tid < off) red[tid] += red[tid + off];
        __syncthreads();
    }
    float inv = 1.f / red[0];
    for (int s = tid; s < Ss; s += 256) {
        float a = arow[s] * inv;
        arow[s] = a;
        pa[b * Ss + s] += a;
    }
}

// c_enc[b,:] = attn[b,:] @ encoder_hy[b,:,:]  (bf16 stream, f32 accumulate).
// Thread handles 2 adjacent e columns -> 32-bit coalesced loads per s.
__global__ void k_cenc(const float* __restrict__ arow_base, const u16* __restrict__ encb,
                       float* __restrict__ cenc, u16* __restrict__ cat2) {
    const int b = blockIdx.x, tid = threadIdx.x;
    __shared__ float s_a[Ss];
    for (int s = tid; s < Ss; s += 256) s_a[s] = arow_base[b * Ss + s];
    __syncthreads();
    const unsigned* eb = (const unsigned*)(encb + (size_t)b * Ss * ENC);
    float a0 = 0.f, a1 = 0.f;
    for (int s = 0; s < Ss; ++s) {
        float a = s_a[s];
        unsigned w = eb[s * (ENC / 2) + tid];
        a0 += a * bf2f((u16)(w & 0xffffu));
        a1 += a * bf2f((u16)(w >> 16));
    }
    int e0 = 2 * tid, e1 = 2 * tid + 1;
    cenc[b * ENC + e0] = a0;
    cenc[b * ENC + e1] = a1;
    cat2[b * (ENC + TRG) + e0] = f2bf(a0);   // c_enc half of [c_enc | h_new]
    cat2[b * (ENC + TRG) + e1] = f2bf(a1);
}

// ha_new = [c_enc | h_new] @ w_ao^T + b_ao; writes ha state and output_[:,t,:].
__global__ void k_ha(const u16* __restrict__ cat2, const u16* __restrict__ waob,
                     const float* __restrict__ bao, float* __restrict__ ha,
                     float* __restrict__ out, int t) {
    int wid = (blockIdx.x * blockDim.x + threadIdx.x) >> 5;
    int mt = wid >> 5, nt = wid & 31;
    int m0 = mt * 16, n0 = nt * 16;
    f32x8 acc = wmma_bf16_AWt(cat2, ENC + TRG, waob, ENC + TRG, m0, n0, ENC + TRG);
    const int lane = threadIdx.x & 31, half = lane >> 4, l16 = lane & 15;
    float bias = bao[n0 + l16];
#pragma unroll
    for (int r = 0; r < 8; ++r) {
        int m = m0 + half * 8 + r;      // m == batch index
        int n = n0 + l16;
        float v = acc[r] + bias;
        ha[m * TRG + n] = v;
        out[((size_t)m * Tt + t) * TRG + n] = v;
    }
}

// p[b] = sigmoid([x_t | h_new | c_enc] . w_pt + b_pt).  Wave per b.
__global__ void k_pgen(const float* __restrict__ input_, const float* __restrict__ h,
                       const float* __restrict__ cenc, const float* __restrict__ wpt,
                       const float* __restrict__ bpt, float* __restrict__ pgen, int t) {
    int wid = (blockIdx.x * blockDim.x + threadIdx.x) >> 5;
    int lane = threadIdx.x & 31;
    int b = wid;
    float acc = 0.f;
    for (int k = lane; k < KCAT; k += 32) {
        float v = (k < INs)       ? input_[(b * Tt + t) * INs + k]
                : (k < INs + TRG) ? h[b * TRG + (k - INs)]
                                  : cenc[b * ENC + (k - INs - TRG)];
        acc += v * wpt[k];
    }
#pragma unroll
    for (int m = 16; m > 0; m >>= 1) acc += __shfl_xor(acc, m);
    if (lane == 0) pgen[b * Tt + t] = sigm(acc + bpt[0]);
}

// ---------------------------------------------------------------------------
// Host launch
// ---------------------------------------------------------------------------

extern "C" void kernel_launch(void* const* d_in, const int* in_sizes, int n_in,
                              void* d_out, int out_size, void* d_ws, size_t ws_size,
                              hipStream_t stream) {
    // Inputs (setup_inputs dict order)
    const float* input_  = (const float*)d_in[1];
    const float* h0      = (const float*)d_in[2];
    const float* c0      = (const float*)d_in[3];
    const float* h_attn  = (const float*)d_in[4];
    const float* enc_hy  = (const float*)d_in[5];
    const float* p_attn  = (const float*)d_in[6];
    const float* p_dehy  = (const float*)d_in[7];
    const float* w_ih    = (const float*)d_in[8];
    const float* b_ih    = (const float*)d_in[9];
    const float* w_hh    = (const float*)d_in[10];
    const float* b_hh    = (const float*)d_in[11];
    const float* w_en    = (const float*)d_in[12];
    const float* b_en    = (const float*)d_in[13];
    const float* w_de    = (const float*)d_in[14];
    const float* w_cv    = (const float*)d_in[15];
    const float* w_warp  = (const float*)d_in[16];
    const float* w_ao    = (const float*)d_in[17];
    const float* b_ao    = (const float*)d_in[18];
    const float* w_pt    = (const float*)d_in[19];
    const float* b_pt    = (const float*)d_in[20];

    // Output slabs (flat, return order)
    float* out      = (float*)d_out;
    float* o_output = out;                                  // [B,T,TRG]
    float* o_hT     = o_output + (size_t)Bb * Tt * TRG;     // [B,TRG]   (h state)
    float* o_cT     = o_hT + Bb * TRG;                      // [B,TRG]   (c state)
    float* o_haT    = o_cT + Bb * TRG;                      // [B,TRG]   (ha state)
    float* o_attns  = o_haT + Bb * TRG;                     // [T,B,S]
    float* o_paT    = o_attns + (size_t)Tt * Bb * Ss;       // [B,S]     (pa state)
    float* o_pgen   = o_paT + Bb * Ss;                      // [B,T]
    float* o_pdehy  = o_pgen + Bb * Tt;                     // [B,1,TRG]
    float* o_loss   = o_pdehy + Bb * TRG;                   // [1]

    // Workspace layout
    char* ws = (char*)d_ws;
    u16* enc_bf  = (u16*)ws;                       ws += (size_t)Bb * Ss * ENC * 2;        // 64 MB
    u16* ep_bf   = (u16*)ws;                       ws += (size_t)Bb * Ss * TRG * 2;        // 64 MB
    u16* wcat    = (u16*)ws;                       ws += (size_t)Gg * KCAT * 2;            // 5 MB
    u16* wenb    = (u16*)ws;                       ws += (size_t)TRG * ENC * 2;
    u16* wdeb    = (u16*)ws;                       ws += (size_t)TRG * TRG * 2;
    u16* waob    = (u16*)ws;                       ws += (size_t)TRG * (ENC + TRG) * 2;
    u16* xcat    = (u16*)ws;                       ws += (size_t)Bb * KCAT * 2;
    u16* hbf     = (u16*)ws;                       ws += (size_t)Bb * TRG * 2;
    u16* cat2    = (u16*)ws;                       ws += (size_t)Bb * (ENC + TRG) * 2;
    float* gates = (float*)ws;                     ws += (size_t)Bb * Gg * 4;
    float* dh    = (float*)ws;                     ws += (size_t)Bb * TRG * 4;
    float* ee    = (float*)ws;                     ws += (size_t)Bb * Ss * 4;
    float* cenc  = (float*)ws;                     ws += (size_t)Bb * ENC * 4;

    // ---- one-time setup ----
    {
        const int n_init = Bb * TRG * 3 + Bb * Ss + Bb * TRG + 1;
        k_init<<<(n_init + 255) / 256, 256, 0, stream>>>(
            h0, c0, h_attn, p_attn, p_dehy, o_hT, o_cT, o_haT, o_paT, o_pdehy, o_loss);

        const int n_conv = Bb * Ss * ENC + Gg * KCAT + TRG * ENC + TRG * TRG + TRG * (ENC + TRG);
        k_convert<<<(n_conv + 255) / 256, 256, 0, stream>>>(
            enc_hy, w_ih, w_hh, w_en, w_de, w_ao, enc_bf, wcat, wenb, wdeb, waob);

        // 4096 x 32 tiles, 8 waves/block -> 16384 blocks
        k_encproj<<<16384, 256, 0, stream>>>(enc_bf, wenb, b_en, ep_bf);
    }

    // ---- 100 sequential decode steps ----
    for (int t = 0; t < Tt; ++t) {
        float* attn_t = o_attns + (size_t)t * Bb * Ss;

        k_stagex<<<(Bb * KCAT + 255) / 256, 256, 0, stream>>>(input_, o_haT, o_hT, xcat, t);
        k_gates <<<64, 256, 0, stream>>>(xcat, wcat, b_ih, b_hh, gates);      // 512 waves
        k_cell  <<<(Bb * TRG + 255) / 256, 256, 0, stream>>>(gates, o_hT, o_cT, hbf, cat2);
        k_dh    <<<16, 256, 0, stream>>>(hbf, wdeb, dh);                      // 128 waves
        k_attnee<<<Bb * (Ss / 8), 256, 0, stream>>>(ep_bf, dh, w_cv, w_warp, o_paT, ee);
        k_softmax<<<Bb, 256, 0, stream>>>(ee, attn_t, o_paT);
        k_cenc  <<<Bb, 256, 0, stream>>>(attn_t, enc_bf, cenc, cat2);
        k_ha    <<<16, 256, 0, stream>>>(cat2, waob, b_ao, o_haT, o_output, t); // 128 waves
        k_pgen  <<<8, 256, 0, stream>>>(input_, o_hT, cenc, w_pt, b_pt, o_pgen, t);
    }
}
